// Propogator_33844342292619
// MI455X (gfx1250) — matrix operand order repaired and checked by reading, not compiled, not run on hardware
//
#include <hip/hip_runtime.h>
#include <hip/hip_bf16.h>

// ---------------------------------------------------------------------------
// GGNN propagator step for MI455X (gfx1250, wave32, WMMA).
//   a_in  = A_in  @ state_in   (scatter-add SpMM, f32 atomics, L2-resident)
//   a_out = A_out @ state_out
//   a = [a_in | a_out | cur];  r = sigma(a@Wr+br); z = sigma(a@Wz+bz)
//   h = tanh([a_in|a_out|r*cur]@Wh+bh);  out = (1-z)*cur + z*h
// GEMMs use v_wmma_f32_16x16x32_bf16 (bf16 in, f32 accumulate).
// MT=32 row tiles: each B (weight) fragment is reused by two row-subtiles,
// halving L2 weight-refetch traffic vs MT=16.
// ---------------------------------------------------------------------------

#define N_NODE  100000
#define D_DIM   128
#define M_COLS  200000
#define NNZ_CNT 3200000
#define K3      384
#define MT      32            // rows per block (N_NODE % MT == 0 -> 3125 blocks)

typedef __bf16 bf16;
typedef __attribute__((ext_vector_type(16))) __bf16 v16bf;
typedef __attribute__((ext_vector_type(8)))  float  v8f;

// ----------------------------- utility kernels -----------------------------

__global__ __launch_bounds__(256) void zero_f4(float4* __restrict__ p) {
    size_t i = (size_t)blockIdx.x * blockDim.x + threadIdx.x;
    p[i] = make_float4(0.f, 0.f, 0.f, 0.f);
}

// Convert the three f32 [384,128] weight matrices to bf16, transposed to
// [col][K] so WMMA B-fragment K-pairs are contiguous 4-byte loads.
__global__ __launch_bounds__(256) void cvt_weights(
        const float* __restrict__ Wr, const float* __restrict__ Wz,
        const float* __restrict__ Wh,
        bf16* __restrict__ Wrt, bf16* __restrict__ Wzt, bf16* __restrict__ Wht) {
    int e = blockIdx.x * blockDim.x + threadIdx.x;   // 0 .. 3*384*128-1
    int mat = e / (K3 * D_DIM);
    int rem = e - mat * (K3 * D_DIM);
    int k = rem / D_DIM;
    int n = rem - k * D_DIM;
    const float* W = (mat == 0) ? Wr : (mat == 1) ? Wz : Wh;
    bf16* Wt = (mat == 0) ? Wrt : (mat == 1) ? Wzt : Wht;
    Wt[(size_t)n * K3 + k] = (bf16)W[rem];
}

// -------------------------------- SpMM -------------------------------------
// One wave per nonzero: scalar (row,col,val), 32 lanes x float4 = 128 floats
// of x[col], 4 f32 atomic adds per lane into a[row]. Sources + accumulators
// are L2-resident (192 MB), so the scatter-add stays on-chip.
__global__ __launch_bounds__(256) void spmm_atomic(
        const int* __restrict__ rows, const int* __restrict__ cols,
        const float* __restrict__ vals, const float* __restrict__ x,
        float* __restrict__ acc) {
    int wave = threadIdx.x >> 5;
    int lane = threadIdx.x & 31;
    size_t e = (size_t)blockIdx.x * 8 + wave;   // NNZ divisible by 8
    int row = rows[e];
    int col = cols[e];
    float v = vals[e];
    float4 xv = ((const float4*)x)[(size_t)col * (D_DIM / 4) + lane];
    float* dst = acc + (size_t)row * D_DIM + lane * 4;
    atomicAdd(dst + 0, v * xv.x);
    atomicAdd(dst + 1, v * xv.y);
    atomicAdd(dst + 2, v * xv.z);
    atomicAdd(dst + 3, v * xv.w);
}

// --------------------------- fragment builders -----------------------------
// ISA 7.12.2 16-bit A 16x32 layout: lanes share M=lane&15, half=lane>>4 picks
// the K group; VGPR v packs K = 16*(v>>2) + 8*half + 2*(v&3) (+1).
__device__ inline v16bf load_a_frag(const bf16* __restrict__ base, int stride,
                                    int m, int half, int kb) {
    v16bf a;
#pragma unroll
    for (int v = 0; v < 8; ++v) {
        int k0 = kb + ((v >> 2) << 4) + (half << 3) + ((v & 3) << 1);
        a[2 * v]     = base[m * stride + k0];
        a[2 * v + 1] = base[m * stride + k0 + 1];
    }
    return a;
}

// B 32x16 layout: lane holds column n=lane&15; half=lane>>4 picks the K
// group; VGPR v packs K = 16*(v>>2) + 8*half + 2*(v&3) (+1). colptr points at Wt[col][0].
__device__ inline v16bf load_b_frag(const bf16* __restrict__ colptr,
                                    int half, int kb) {
    v16bf b;
#pragma unroll
    for (int v = 0; v < 8; ++v) {
        int k = kb + ((v >> 2) << 4) + (half << 3) + ((v & 3) << 1);
        b[2 * v]     = colptr[k];
        b[2 * v + 1] = colptr[k + 1];
    }
    return b;
}

__device__ inline float sigmoidf_(float x) {
    return 1.f / (1.f + __expf(-x));
}

// ----------------------------- fused gate GEMM -----------------------------
// MT=32 rows per block; 8 waves; wave w owns output columns 16w..16w+15 and
// two 16-row subtiles. Each B fragment loaded once per K-step is shared by
// both subtiles (halves weight refetch vs MT=16).
// Phase 1: K 0..255   -> accR, accZ, accH
// Phase 2: K 256..383 -> accR, accZ (cur part)
// Phase 3: r*cur via LDS -> 4 WMMA steps finish accH; GRU blend; store.
#define SA_STRIDE 392   // 384 + pad
#define SR_STRIDE 136   // 128 + pad

__global__ __launch_bounds__(256) void ggnn_gates(
        const float* __restrict__ a_in, const float* __restrict__ a_out,
        const float* __restrict__ cur,
        const bf16* __restrict__ Wrt, const bf16* __restrict__ Wzt,
        const bf16* __restrict__ Wht,
        const float* __restrict__ br, const float* __restrict__ bz,
        const float* __restrict__ bh, float* __restrict__ out) {
    __shared__ bf16 sA[MT * SA_STRIDE];   // [a_in|a_out|cur] tile, bf16 (~25KB)
    __shared__ bf16 sRC[MT * SR_STRIDE];  // r*cur tile, bf16 (~8.5KB)

    const int tid  = threadIdx.x;
    const int wave = tid >> 5;
    const int lane = tid & 31;
    const int half = lane >> 4;
    const int ln16 = lane & 15;
    const size_t rowbase = (size_t)blockIdx.x * MT;   // N divisible by MT

    // Stage the MTx384 activation tile (f32 -> bf16), coalesced over k.
    for (int e = tid; e < MT * K3; e += 256) {
        int rr = e / K3;
        int k  = e - rr * K3;
        size_t grow = rowbase + rr;
        float v;
        if (k < 128)      v = a_in[grow * D_DIM + k];
        else if (k < 256) v = a_out[grow * D_DIM + (k - 128)];
        else              v = cur[grow * D_DIM + (k - 256)];
        sA[rr * SA_STRIDE + k] = (bf16)v;
    }
    // Speculative prefetch of the next tile (gfx1250 global_prefetch_b8).
    if (rowbase + MT < N_NODE)
        __builtin_prefetch(&a_in[(rowbase + MT) * D_DIM + (tid & 127)], 0, 1);
    __syncthreads();

    const int col = (wave << 4) + ln16;               // 0..127
    const bf16* wr = Wrt + (size_t)col * K3;
    const bf16* wz = Wzt + (size_t)col * K3;
    const bf16* wh = Wht + (size_t)col * K3;

    v8f accR[2], accZ[2], accH[2];
#pragma unroll
    for (int s = 0; s < 2; ++s) {
        accR[s] = (v8f)(0.0f);
        accZ[s] = (v8f)(0.0f);
        accH[s] = (v8f)(0.0f);
    }

    // Phase 1: K = 0..255 ([a_in|a_out]) -> r, z, h-partial
    for (int kb = 0; kb < 256; kb += 32) {
        v16bf bR = load_b_frag(wr, half, kb);
        v16bf bZ = load_b_frag(wz, half, kb);
        v16bf bH = load_b_frag(wh, half, kb);
#pragma unroll
        for (int s = 0; s < 2; ++s) {
            v16bf af = load_a_frag(sA + s * 16 * SA_STRIDE, SA_STRIDE,
                                   ln16, half, kb);
            accR[s] = __builtin_amdgcn_wmma_f32_16x16x32_bf16(
                false, af, false, bR, (short)0, accR[s], false, false);
            accZ[s] = __builtin_amdgcn_wmma_f32_16x16x32_bf16(
                false, af, false, bZ, (short)0, accZ[s], false, false);
            accH[s] = __builtin_amdgcn_wmma_f32_16x16x32_bf16(
                false, af, false, bH, (short)0, accH[s], false, false);
        }
    }
    // Phase 2: K = 256..383 (cur) -> r, z only
    for (int kb = 256; kb < 384; kb += 32) {
        v16bf bR = load_b_frag(wr, half, kb);
        v16bf bZ = load_b_frag(wz, half, kb);
#pragma unroll
        for (int s = 0; s < 2; ++s) {
            v16bf af = load_a_frag(sA + s * 16 * SA_STRIDE, SA_STRIDE,
                                   ln16, half, kb);
            accR[s] = __builtin_amdgcn_wmma_f32_16x16x32_bf16(
                false, af, false, bR, (short)0, accR[s], false, false);
            accZ[s] = __builtin_amdgcn_wmma_f32_16x16x32_bf16(
                false, af, false, bZ, (short)0, accZ[s], false, false);
        }
    }

    // Gates. C layout (7.12.2): reg p -> row m = p + 8*half, col n = ln16.
    const float brc = br[col], bzc = bz[col], bhc = bh[col];
    float zv[2][8], cv[2][8];
#pragma unroll
    for (int s = 0; s < 2; ++s) {
#pragma unroll
        for (int p = 0; p < 8; ++p) {
            int m = s * 16 + p + (half << 3);
            size_t grow = rowbase + m;
            float c = cur[grow * D_DIM + col];
            cv[s][p] = c;
            float r  = sigmoidf_(accR[s][p] + brc);
            zv[s][p] = sigmoidf_(accZ[s][p] + bzc);
            sRC[m * SR_STRIDE + col] = (bf16)(r * c);
        }
    }
    __syncthreads();

    // Phase 3: (r*cur) @ Wh[256:384] from LDS.
    for (int kb = 0; kb < 128; kb += 32) {
        v16bf bH = load_b_frag(wh + 256, half, kb);
#pragma unroll
        for (int s = 0; s < 2; ++s) {
            v16bf af = load_a_frag(sRC + s * 16 * SR_STRIDE, SR_STRIDE,
                                   ln16, half, kb);
            accH[s] = __builtin_amdgcn_wmma_f32_16x16x32_bf16(
                false, af, false, bH, (short)0, accH[s], false, false);
        }
    }

#pragma unroll
    for (int s = 0; s < 2; ++s) {
#pragma unroll
        for (int p = 0; p < 8; ++p) {
            int m = s * 16 + p + (half << 3);
            size_t grow = rowbase + m;
            float h = tanhf(accH[s][p] + bhc);
            out[grow * D_DIM + col] = (1.f - zv[s][p]) * cv[s][p] + zv[s][p] * h;
        }
    }
}

// ------------------------------- launcher ----------------------------------

extern "C" void kernel_launch(void* const* d_in, const int* in_sizes, int n_in,
                              void* d_out, int out_size, void* d_ws, size_t ws_size,
                              hipStream_t stream) {
    const float* state_in  = (const float*)d_in[0];
    const float* state_out = (const float*)d_in[1];
    const float* state_cur = (const float*)d_in[2];
    const int*   Ain_r  = (const int*)d_in[3];
    const int*   Ain_c  = (const int*)d_in[4];
    const float* Ain_v  = (const float*)d_in[5];
    const int*   Aout_r = (const int*)d_in[6];
    const int*   Aout_c = (const int*)d_in[7];
    const float* Aout_v = (const float*)d_in[8];
    const float* Wr = (const float*)d_in[9];
    const float* br = (const float*)d_in[10];
    const float* Wz = (const float*)d_in[11];
    const float* bz = (const float*)d_in[12];
    const float* Wh = (const float*)d_in[13];
    const float* bh = (const float*)d_in[14];
    float* out = (float*)d_out;

    // Workspace: a_in (51.2MB) | a_out (51.2MB) | Wrt/Wzt/Wht bf16 (96KB each)
    float* a_in  = (float*)d_ws;
    float* a_out = a_in + (size_t)N_NODE * D_DIM;
    bf16*  Wrt   = (bf16*)(a_out + (size_t)N_NODE * D_DIM);
    bf16*  Wzt   = Wrt + K3 * D_DIM;
    bf16*  Wht   = Wzt + K3 * D_DIM;

    // 1) zero both accumulators (contiguous: 2*N*D floats = 6.4M float4)
    zero_f4<<<(2 * N_NODE * D_DIM / 4) / 256, 256, 0, stream>>>((float4*)a_in);

    // 2) bf16 weight transpose (3*384*128 = 147456 elements)
    cvt_weights<<<(3 * K3 * D_DIM) / 256, 256, 0, stream>>>(Wr, Wz, Wh, Wrt, Wzt, Wht);

    // 3) SpMMs: one wave per nonzero, 8 nnz per 256-thread block
    spmm_atomic<<<NNZ_CNT / 8, 256, 0, stream>>>(Ain_r, Ain_c, Ain_v, state_in, a_in);
    spmm_atomic<<<NNZ_CNT / 8, 256, 0, stream>>>(Aout_r, Aout_c, Aout_v, state_out, a_out);

    // 4) fused WMMA gate GEMM + GRU update (N/MT = 3125 row tiles)
    ggnn_gates<<<N_NODE / MT, 256, 0, stream>>>(a_in, a_out, state_cur,
                                                Wrt, Wzt, Wht, br, bz, bh, out);
}
